// SlotAttention_42537356100060
// MI455X (gfx1250) — matrix operand order repaired and checked by reading, not compiled, not run on hardware
//
#include <hip/hip_runtime.h>
#include <hip/hip_bf16.h>
#include <math.h>

// ---------------------------------------------------------------------------
// Slot Attention on MI455X (gfx1250), bf16 WMMA GEMMs + fp32 elementwise.
// ---------------------------------------------------------------------------

typedef __bf16 bf16;
typedef __attribute__((ext_vector_type(16))) __bf16 v16bf;
typedef __attribute__((ext_vector_type(8)))  float  v8f;

union BF16x16 { v16bf v; int4 q[2]; };

#define CB   32      // batch
#define CN   2048    // tokens
#define CD   512     // dim
#define CNS  8       // slots
#define CH   512     // mlp hidden
#define CIT  3
#define CEPS 1e-8f
#define CLNE 1e-5f

// ---------------------------------------------------------------------------
// Generic GEMM:  C[M,N] = alpha * A[M,K] x W[N,K]^T (+bias)(+residual)(relu?)
// A,W bf16 row-major contiguous in K.  Outputs: f32 row-major, bf16 row-major,
// and/or bf16 transposed-per-sub-batch (CbT[row/tRows][col][row%tRows]).
// Block = 256 threads = 8 waves (2 along M x 4 along N); block tile 32x256;
// wave tile 16x64 (4 accumulators).  A tile staged in LDS (dynamic).
// Requires: K % 32 == 0, N % 256 == 0. M guarded (padded to 32 internally).
// ---------------------------------------------------------------------------
__global__ __launch_bounds__(256)
void gemm_nt_k(const bf16* __restrict__ A, long sA,
               const bf16* __restrict__ W, long sW,
               const float* __restrict__ bias,
               const float* __restrict__ residual, long sR,
               float* __restrict__ Cf, long sCf,
               bf16* __restrict__ Cb, long sCb,
               bf16* __restrict__ CbT, int tRows,
               int M, int N, int K, float alpha, int relu)
{
    extern __shared__ bf16 As[];       // [32][K+8]
    const int ldA = K + 8;
    const int tid = threadIdx.x;
    const int bz  = blockIdx.z;
    const int bm  = blockIdx.x * 32;
    const int bn  = blockIdx.y * 256;

    const bf16* Ab = A + (long)bz * sA;
    const bf16* Wb = W + (long)bz * sW;

    // ---- stage A tile (32 rows x K) into LDS, zero-fill out-of-range rows
    const int KC = K >> 3;             // 8 bf16 (16B) chunks per row
    for (int c = tid; c < 32 * KC; c += 256) {
        int r  = c / KC;
        int kc = c - r * KC;
        int4 val = make_int4(0, 0, 0, 0);
        int grow = bm + r;
        if (grow < M) val = *(const int4*)(Ab + (long)grow * K + (kc << 3));
        *(int4*)(As + r * ldA + (kc << 3)) = val;
    }
    __syncthreads();

    const int wave = tid >> 5, lane = tid & 31;
    const int wm = wave & 1;           // 2 waves along M
    const int wn = wave >> 1;          // 4 waves along N
    const int m  = lane & 15;
    const int kh = lane >> 4;

    const bf16* arow   = As + (wm * 16 + m) * ldA;
    const int  colBase = bn + wn * 64 + m;

    v8f acc[4] = {};
    for (int kk = 0; kk < K; kk += 32) {
        BF16x16 af;
        af.q[0] = *(const int4*)(arow + kk + kh * 8);
        af.q[1] = *(const int4*)(arow + kk + kh * 8 + 16);
        const bf16* wp = Wb + (long)colBase * K + kk + kh * 8;
#pragma unroll
        for (int t = 0; t < 4; t++) {
            BF16x16 bfm;
            bfm.q[0] = *(const int4*)(wp);
            bfm.q[1] = *(const int4*)(wp + 16);
            wp += (long)16 * K;
            acc[t] = __builtin_amdgcn_wmma_f32_16x16x32_bf16(
                false, af.v, false, bfm.v, (short)0, acc[t], false, false);
        }
    }

    // ---- epilogue / stores
    const int grow0 = bm + wm * 16 + kh * 8;
#pragma unroll
    for (int t = 0; t < 4; t++) {
        int col = colBase + t * 16;
        float bv = bias ? bias[col] : 0.0f;
        float vals[8];
#pragma unroll
        for (int r = 0; r < 8; r++) {
            float x = acc[t][r] * alpha + bv;
            if (residual && (grow0 + r) < M)
                x += residual[(long)bz * sR + (long)(grow0 + r) * N + col];
            if (relu) x = fmaxf(x, 0.0f);
            vals[r] = x;
        }
        if (Cf) {
            float* o = Cf + (long)bz * sCf;
#pragma unroll
            for (int r = 0; r < 8; r++)
                if ((grow0 + r) < M) o[(long)(grow0 + r) * N + col] = vals[r];
        }
        if (Cb) {
            bf16* o = Cb + (long)bz * sCb;
#pragma unroll
            for (int r = 0; r < 8; r++)
                if ((grow0 + r) < M) o[(long)(grow0 + r) * N + col] = (bf16)vals[r];
        }
        if (CbT) {   // packed 16B transposed store (rows assumed in-range)
            union { int4 q; bf16 h[8]; } pk;
#pragma unroll
            for (int r = 0; r < 8; r++) pk.h[r] = (bf16)vals[r];
            int b2 = grow0 / tRows;
            int j0 = grow0 - b2 * tRows;
            *(int4*)(CbT + (long)b2 * ((long)N * tRows) + (long)col * tRows + j0) = pk.q;
        }
    }
}

// ---------------------------------------------------------------------------
// slots = mu + exp(logsigma) * noise        (n = B*NS*D elements)
// ---------------------------------------------------------------------------
__global__ void init_slots_k(const float* __restrict__ mu,
                             const float* __restrict__ ls,
                             const float* __restrict__ noise,
                             float* __restrict__ slots, int n)
{
    int i = blockIdx.x * 256 + threadIdx.x;
    if (i >= n) return;
    int d = i & (CD - 1);
    slots[i] = mu[d] + expf(ls[d]) * noise[i];
}

// ---------------------------------------------------------------------------
// f32 -> bf16 convert
// ---------------------------------------------------------------------------
__global__ void cvt_bf16_k(const float* __restrict__ in, bf16* __restrict__ out, int n)
{
    int i = blockIdx.x * 256 + threadIdx.x;
    if (i < n) out[i] = (bf16)in[i];
}

// ---------------------------------------------------------------------------
// LayerNorm over last dim (=512) -> bf16.  One wave32 per row.
// ---------------------------------------------------------------------------
__global__ void ln_rows_k(const float* __restrict__ x,
                          const float* __restrict__ g,
                          const float* __restrict__ b,
                          bf16* __restrict__ out, int rows)
{
    int gid  = blockIdx.x * blockDim.x + threadIdx.x;
    int row  = gid >> 5;
    int lane = gid & 31;
    if (row >= rows) return;
    const float* xr = x + (long)row * CD;
    float v[16];
    float s = 0.0f;
#pragma unroll
    for (int t = 0; t < 16; t++) { v[t] = xr[lane + 32 * t]; s += v[t]; }
#pragma unroll
    for (int o = 16; o >= 1; o >>= 1) s += __shfl_xor(s, o, 32);
    float mean = s * (1.0f / CD);
    float q = 0.0f;
#pragma unroll
    for (int t = 0; t < 16; t++) { float d = v[t] - mean; q += d * d; }
#pragma unroll
    for (int o = 16; o >= 1; o >>= 1) q += __shfl_xor(q, o, 32);
    float rstd = rsqrtf(q * (1.0f / CD) + CLNE);
    bf16* orow = out + (long)row * CD;
#pragma unroll
    for (int t = 0; t < 16; t++) {
        int c = lane + 32 * t;
        orow[c] = (bf16)((v[t] - mean) * rstd * g[c] + b[c]);
    }
}

// ---------------------------------------------------------------------------
// softmax over slots axis (NS=8) per (b, key j) column
// ---------------------------------------------------------------------------
__global__ void softmax_slots_k(const float* __restrict__ dots, float* __restrict__ p)
{
    int idx = blockIdx.x * 256 + threadIdx.x;   // over B*N columns
    if (idx >= CB * CN) return;
    int b = idx >> 11;           // / 2048
    int j = idx & (CN - 1);
    const float* dp = dots + (long)b * CNS * CN + j;
    float mx = dp[0];
#pragma unroll
    for (int i = 1; i < CNS; i++) mx = fmaxf(mx, dp[(long)i * CN]);
    float e[CNS]; float s = 0.0f;
#pragma unroll
    for (int i = 0; i < CNS; i++) { e[i] = expf(dp[(long)i * CN] - mx); s += e[i]; }
    float inv = 1.0f / s;
    float* pp = p + (long)b * CNS * CN + j;
#pragma unroll
    for (int i = 0; i < CNS; i++) pp[(long)i * CN] = e[i] * inv;
}

// ---------------------------------------------------------------------------
// attn = (p + EPS) / sum_j(p + EPS)  -> bf16.  One block (256 thr) per row.
// ---------------------------------------------------------------------------
__global__ void attn_renorm_k(const float* __restrict__ p, bf16* __restrict__ attn)
{
    __shared__ float sd[256];
    int row = blockIdx.x;                   // b*NS + i
    int tid = threadIdx.x;
    const float* pr = p + (long)row * CN;
    float s = 0.0f;
    for (int j = tid; j < CN; j += 256) s += pr[j];
    sd[tid] = s;
    __syncthreads();
    for (int o = 128; o >= 1; o >>= 1) {
        if (tid < o) sd[tid] += sd[tid + o];
        __syncthreads();
    }
    float inv = 1.0f / (sd[0] + CEPS * (float)CN);
    bf16* ar = attn + (long)row * CN;
    for (int j = tid; j < CN; j += 256) ar[j] = (bf16)((pr[j] + CEPS) * inv);
}

// ---------------------------------------------------------------------------
// GRU gates: slots = (1-z)*n + z*slots  (gate order r,z,n)
// ---------------------------------------------------------------------------
__global__ void gru_k(const float* __restrict__ gi, const float* __restrict__ gh,
                      float* __restrict__ slots)
{
    int i = blockIdx.x * 256 + threadIdx.x;
    if (i >= CB * CNS * CD) return;
    int row = i >> 9;            // / 512
    int d   = i & (CD - 1);
    const float* gir = gi + (long)row * (3 * CD);
    const float* ghr = gh + (long)row * (3 * CD);
    float r = 1.0f / (1.0f + expf(-(gir[d] + ghr[d])));
    float z = 1.0f / (1.0f + expf(-(gir[CD + d] + ghr[CD + d])));
    float n = tanhf(gir[2 * CD + d] + r * ghr[2 * CD + d]);
    slots[i] = (1.0f - z) * n + z * slots[i];
}

// ---------------------------------------------------------------------------
// host side
// ---------------------------------------------------------------------------
static void launch_gemm(hipStream_t stream,
                        const bf16* A, long sA, const bf16* W, long sW,
                        const float* bias, const float* res, long sR,
                        float* Cf, long sCf, bf16* Cb, long sCb,
                        bf16* CbT, int tRows,
                        int M, int N, int K, float alpha, int relu, int batch)
{
    dim3 grid((M + 31) / 32, N / 256, batch);
    size_t sm = (size_t)32 * (K + 8) * sizeof(bf16);
    gemm_nt_k<<<grid, dim3(256), sm, stream>>>(A, sA, W, sW, bias, res, sR,
                                               Cf, sCf, Cb, sCb, CbT, tRows,
                                               M, N, K, alpha, relu);
}

extern "C" void kernel_launch(void* const* d_in, const int* in_sizes, int n_in,
                              void* d_out, int out_size, void* d_ws, size_t ws_size,
                              hipStream_t stream)
{
    (void)in_sizes; (void)n_in; (void)out_size; (void)ws_size;
    const float* inputs   = (const float*)d_in[0];
    const float* noise    = (const float*)d_in[1];
    const float* mu       = (const float*)d_in[2];
    const float* lsg      = (const float*)d_in[3];
    const float* Wq       = (const float*)d_in[4];
    const float* bq       = (const float*)d_in[5];
    const float* Wk       = (const float*)d_in[6];
    const float* bk       = (const float*)d_in[7];
    const float* Wv       = (const float*)d_in[8];
    const float* bv       = (const float*)d_in[9];
    const float* W_ih     = (const float*)d_in[10];
    const float* b_ih     = (const float*)d_in[11];
    const float* W_hh     = (const float*)d_in[12];
    const float* b_hh     = (const float*)d_in[13];
    const float* W1       = (const float*)d_in[14];
    const float* b1       = (const float*)d_in[15];
    const float* W2       = (const float*)d_in[16];
    const float* b2       = (const float*)d_in[17];
    const float* ln_in_g  = (const float*)d_in[18];
    const float* ln_in_b  = (const float*)d_in[19];
    const float* ln_s_g   = (const float*)d_in[20];
    const float* ln_s_b   = (const float*)d_in[21];
    const float* ln_ff_g  = (const float*)d_in[22];
    const float* ln_ff_b  = (const float*)d_in[23];

    // workspace bump allocator
    char* ws = (char*)d_ws;
    size_t off = 0;
    auto alloc = [&](size_t bytes) -> char* {
        char* p = ws + off;
        off = (off + bytes + 255) & ~(size_t)255;
        return p;
    };

    const long MBIG = (long)CB * CN;   // 65536 rows
    bf16* xb    = (bf16*)alloc(MBIG * CD * 2);
    bf16* kb    = (bf16*)alloc(MBIG * CD * 2);
    bf16* vT    = (bf16*)alloc(MBIG * CD * 2);      // [B][D][N]
    bf16* wqb   = (bf16*)alloc((size_t)CD * CD * 2);
    bf16* wkb   = (bf16*)alloc((size_t)CD * CD * 2);
    bf16* wvb   = (bf16*)alloc((size_t)CD * CD * 2);
    bf16* wihb  = (bf16*)alloc((size_t)3 * CD * CD * 2);
    bf16* whhb  = (bf16*)alloc((size_t)3 * CD * CD * 2);
    bf16* w1b   = (bf16*)alloc((size_t)CH * CD * 2);
    bf16* w2b   = (bf16*)alloc((size_t)CD * CH * 2);
    float* slots = (float*)alloc((size_t)CB * CNS * CD * 4);
    bf16* slotsb = (bf16*)alloc((size_t)CB * CNS * CD * 2);
    bf16* snormb = (bf16*)alloc((size_t)CB * CNS * CD * 2);
    bf16* qb     = (bf16*)alloc((size_t)CB * CNS * CD * 2);
    float* dots  = (float*)alloc((size_t)CB * CNS * CN * 4);
    float* psm   = (float*)alloc((size_t)CB * CNS * CN * 4);
    bf16* attnb  = (bf16*)alloc((size_t)CB * CNS * CN * 2);
    bf16* updb   = (bf16*)alloc((size_t)CB * CNS * CD * 2);
    float* gi    = (float*)alloc((size_t)CB * CNS * 3 * CD * 4);
    float* gh    = (float*)alloc((size_t)CB * CNS * 3 * CD * 4);
    bf16* ffb    = (bf16*)alloc((size_t)CB * CNS * CD * 2);
    bf16* h1b    = (bf16*)alloc((size_t)CB * CNS * CH * 2);

    auto cvt = [&](const float* src, bf16* dst, int n) {
        cvt_bf16_k<<<(n + 255) / 256, 256, 0, stream>>>(src, dst, n);
    };

    // weights -> bf16
    cvt(Wq, wqb, CD * CD);
    cvt(Wk, wkb, CD * CD);
    cvt(Wv, wvb, CD * CD);
    cvt(W_ih, wihb, 3 * CD * CD);
    cvt(W_hh, whhb, 3 * CD * CD);
    cvt(W1, w1b, CH * CD);
    cvt(W2, w2b, CD * CH);

    // slots = mu + exp(ls)*noise
    {
        int n = CB * CNS * CD;
        init_slots_k<<<(n + 255) / 256, 256, 0, stream>>>(mu, lsg, noise, slots, n);
    }

    // x = LN(inputs) -> bf16
    {
        int rows = CB * CN;
        ln_rows_k<<<(rows * 32 + 255) / 256, 256, 0, stream>>>(inputs, ln_in_g, ln_in_b, xb, rows);
    }

    // k = x Wk^T + bk (bf16 row-major); v = x Wv^T + bv (bf16 transposed per batch)
    launch_gemm(stream, xb, 0, wkb, 0, bk, nullptr, 0,
                nullptr, 0, kb, 0, nullptr, 0,
                (int)MBIG, CD, CD, 1.0f, 0, 1);
    launch_gemm(stream, xb, 0, wvb, 0, bv, nullptr, 0,
                nullptr, 0, nullptr, 0, vT, CN,
                (int)MBIG, CD, CD, 1.0f, 0, 1);

    const float SCALE = 1.0f / sqrtf((float)CD);
    const int SROWS = CB * CNS;      // 256

    for (int it = 0; it < CIT; it++) {
        // slots_prev -> bf16 (for gh GEMM)
        cvt(slots, slotsb, SROWS * CD);

        // s_norm = LN(slots)
        ln_rows_k<<<(SROWS * 32 + 255) / 256, 256, 0, stream>>>(slots, ln_s_g, ln_s_b, snormb, SROWS);

        // q = s_norm Wq^T + bq
        launch_gemm(stream, snormb, 0, wqb, 0, bq, nullptr, 0,
                    nullptr, 0, qb, 0, nullptr, 0,
                    SROWS, CD, CD, 1.0f, 0, 1);

        // dots[b] = SCALE * q[b] k[b]^T  (batched, M=8 guarded)
        launch_gemm(stream, qb, (long)CNS * CD, kb, (long)CN * CD, nullptr, nullptr, 0,
                    dots, (long)CNS * CN, nullptr, 0, nullptr, 0,
                    CNS, CN, CD, SCALE, 0, CB);

        // softmax over slots, then renorm over keys -> attn bf16
        softmax_slots_k<<<(CB * CN + 255) / 256, 256, 0, stream>>>(dots, psm);
        attn_renorm_k<<<SROWS, 256, 0, stream>>>(psm, attnb);

        // updates[b] = attn[b] v[b]   (W = vT[b] is [D,N] row-major)
        launch_gemm(stream, attnb, (long)CNS * CN, vT, (long)CD * CN, nullptr, nullptr, 0,
                    nullptr, 0, updb, (long)CNS * CD, nullptr, 0,
                    CNS, CD, CN, 1.0f, 0, CB);

        // GRU: gi = updates W_ih^T + b_ih ; gh = slots_prev W_hh^T + b_hh
        launch_gemm(stream, updb, 0, wihb, 0, b_ih, nullptr, 0,
                    gi, 0, nullptr, 0, nullptr, 0,
                    SROWS, 3 * CD, CD, 1.0f, 0, 1);
        launch_gemm(stream, slotsb, 0, whhb, 0, b_hh, nullptr, 0,
                    gh, 0, nullptr, 0, nullptr, 0,
                    SROWS, 3 * CD, CD, 1.0f, 0, 1);
        {
            int n = SROWS * CD;
            gru_k<<<(n + 255) / 256, 256, 0, stream>>>(gi, gh, slots);
        }

        // MLP: slots += relu(LN(slots) W1^T + b1) W2^T + b2
        ln_rows_k<<<(SROWS * 32 + 255) / 256, 256, 0, stream>>>(slots, ln_ff_g, ln_ff_b, ffb, SROWS);
        launch_gemm(stream, ffb, 0, w1b, 0, b1, nullptr, 0,
                    nullptr, 0, h1b, 0, nullptr, 0,
                    SROWS, CH, CD, 1.0f, 1, 1);
        launch_gemm(stream, h1b, 0, w2b, 0, b2, slots, 0,
                    slots, 0, nullptr, 0, nullptr, 0,
                    SROWS, CD, CH, 1.0f, 0, 1);
    }

    hipMemcpyAsync(d_out, slots, (size_t)CB * CNS * CD * sizeof(float),
                   hipMemcpyDeviceToDevice, stream);
}
